// TransformerLayer_27693949124970
// MI455X (gfx1250) — matrix operand order repaired and checked by reading, not compiled, not run on hardware
//
#include <hip/hip_runtime.h>

// ---------------- problem dims ----------------
constexpr int S_ = 1024, B_ = 8, D_ = 1024, H_ = 8, E_ = 16;
constexpr int HD_ = D_ / H_;   // 128
constexpr int M_  = S_ * B_;   // 8192 tokens
constexpr float EPS_ = 1e-5f;

typedef __attribute__((ext_vector_type(16))) __bf16 v16bf;
typedef __attribute__((ext_vector_type(8)))  float  v8f;
typedef unsigned short ushort_t;

union Frag { v16bf v; uint4 q[2]; };

__device__ __forceinline__ ushort_t f2bf(float f) {
  unsigned u = __builtin_bit_cast(unsigned, f);
  unsigned r = (u + 0x7FFFu + ((u >> 16) & 1u)) >> 16;   // RNE
  return (ushort_t)r;
}
__device__ __forceinline__ float bf2f(ushort_t h) {
  unsigned u = ((unsigned)h) << 16;
  return __builtin_bit_cast(float, u);
}
__device__ __forceinline__ v8f wmma_bf16(v16bf a, v16bf b, v8f c) {
  return __builtin_amdgcn_wmma_f32_16x16x32_bf16(false, a, false, b, (short)0, c, false, false);
}

// ---------------- fp32 -> bf16 conversion ----------------
__global__ void cvt_f32_bf16(const float* __restrict__ src, ushort_t* __restrict__ dst, int n) {
  int i = blockIdx.x * 256 + threadIdx.x;
  if (i < n) dst[i] = f2bf(src[i]);
}

// ---------------- generic WMMA GEMM: C[m,n] = sum_k A[m,k]*W[n,k] + bias[n] ----------------
// Wave tile: 32(M) x 64(N); B fragments reused across two A fragments; register
// double-buffering pipelines next-K loads behind current WMMAs.
// MODE 0: bf16 out (QKV).  MODE 1: f32 out (O-proj).  MODE 2: gathered expert rows, f32 out.
template <int MODE>
__global__ __launch_bounds__(256) void gemm_wmma(
    const ushort_t* __restrict__ A, const ushort_t* __restrict__ W,
    const float* __restrict__ bias, void* __restrict__ Cout,
    const int* __restrict__ rowlist, const int* __restrict__ cnt,
    const int* __restrict__ offs)
{
  const int lane = threadIdx.x & 31;
  const int wv   = threadIdx.x >> 5;     // 8 waves = 8 column slices
  const int half = lane >> 4;            // lanes 0-15 / 16-31
  const int col  = lane & 15;
  const int n0 = blockIdx.x * 512 + wv * 64;
  const int m0 = blockIdx.y * 32;

  int e = 0, entry0 = 0, entry1 = 0, valid0 = 0, valid1 = 0;
  size_t abase0, abase1;
  const ushort_t* Wp = W;
  if constexpr (MODE == 2) {
    e = blockIdx.z;
    const int cnte = cnt[e];
    if (m0 >= cnte) return;              // block-uniform early exit
    const int p0 = m0 + col, p1 = m0 + 16 + col;
    valid0 = (p0 < cnte) ? 1 : 0;
    valid1 = (p1 < cnte) ? 1 : 0;
    entry0 = valid0 ? rowlist[offs[e] + p0] : 2 * M_;   // dummy row for tails
    entry1 = valid1 ? rowlist[offs[e] + p1] : 2 * M_;
    abase0 = (size_t)(entry0 >> 1) * D_;
    abase1 = (size_t)(entry1 >> 1) * D_;
    Wp = W + (size_t)e * D_ * D_;
  } else {
    abase0 = (size_t)(m0 + col) * D_;
    abase1 = (size_t)(m0 + 16 + col) * D_;
    (void)rowlist; (void)cnt; (void)offs;
  }

  auto loadA = [&](size_t abase, int kk) {
    Frag f;
    const ushort_t* p = A + abase + kk + half * 8;
    f.q[0] = *(const uint4*)(p);
    f.q[1] = *(const uint4*)(p + 16);
    return f;
  };
  auto loadB = [&](int j, int kk) {
    Frag f;
    const ushort_t* p = Wp + (size_t)(n0 + j * 16 + col) * D_ + kk + half * 8;
    f.q[0] = *(const uint4*)(p);
    f.q[1] = *(const uint4*)(p + 16);
    return f;
  };

  v8f acc[8] = {v8f{}, v8f{}, v8f{}, v8f{}, v8f{}, v8f{}, v8f{}, v8f{}};
  Frag a0c = loadA(abase0, 0);
  Frag a1c = loadA(abase1, 0);
  Frag bc  = loadB(0, 0);
  for (int kk = 0; kk < D_; kk += 32) {
    const int kn = (kk + 32 < D_) ? kk + 32 : kk;
    __builtin_prefetch(A + abase0 + ((kk + 64 < D_) ? kk + 64 : kk), 0, 0);
    Frag a0n = loadA(abase0, kn);        // next-K A frags in flight during WMMAs
    Frag a1n = loadA(abase1, kn);
#pragma unroll
    for (int j = 0; j < 4; ++j) {
      Frag bn = (j < 3) ? loadB(j + 1, kk) : loadB(0, kn);   // next B in flight
      acc[2 * j]     = wmma_bf16(a0c.v, bc.v, acc[2 * j]);
      acc[2 * j + 1] = wmma_bf16(a1c.v, bc.v, acc[2 * j + 1]);
      bc = bn;
    }
    a0c = a0n; a1c = a1n;
  }

#pragma unroll
  for (int j = 0; j < 4; ++j) {
    const int n = n0 + j * 16 + col;
    const float bn = (MODE == 2) ? bias[(size_t)e * D_ + n] : bias[n];
#pragma unroll
    for (int v = 0; v < 8; ++v) {
      const int row = v + half * 8;
      const float cv0 = acc[2 * j][v] + bn;
      const float cv1 = acc[2 * j + 1][v] + bn;
      if constexpr (MODE == 0) {
        ((ushort_t*)Cout)[(size_t)(m0 + row) * D_ + n]      = f2bf(cv0);
        ((ushort_t*)Cout)[(size_t)(m0 + 16 + row) * D_ + n] = f2bf(cv1);
      } else if constexpr (MODE == 1) {
        ((float*)Cout)[(size_t)(m0 + row) * D_ + n]      = cv0;
        ((float*)Cout)[(size_t)(m0 + 16 + row) * D_ + n] = cv1;
      } else {
        int er = __shfl(entry0, row, 32), vl = __shfl(valid0, row, 32);
        if (vl) ((float*)Cout)[(size_t)er * D_ + n] = cv0;
        er = __shfl(entry1, row, 32); vl = __shfl(valid1, row, 32);
        if (vl) ((float*)Cout)[(size_t)er * D_ + n] = cv1;
      }
    }
  }
}

// ---------------- V transpose: [S,B,H,HD] -> [B,H,HD,S] (bf16) ----------------
__global__ void transpose_v(const ushort_t* __restrict__ vb, ushort_t* __restrict__ vt) {
  int o = blockIdx.x * 256 + threadIdx.x;       // over B*H*HD*S = 8M
  int s  = o & (S_ - 1);
  int r  = o >> 10;
  int hd = r & (HD_ - 1);
  int r2 = r >> 7;
  int h  = r2 & (H_ - 1);
  int b  = r2 >> 3;
  vt[o] = vb[(size_t)(s * B_ + b) * D_ + h * HD_ + hd];
}

// ---------------- flash attention ----------------
// Block = 4 waves, all on the same (b,h); K/V tiles staged cooperatively into
// LDS once per 32-key step (4x less global traffic), WMMA B-fragments come from
// ds_load_b128. Rows padded to kill bank conflicts (K: 68-dword stride -> bank
// stride 4; V: 20-dword stride -> all 16 lanes distinct banks).
constexpr int KP_ = 136;   // padded K-tile row (elements)
constexpr int VP_ = 40;    // padded V-tile row (elements)

__global__ __launch_bounds__(128) void flash_attn(
    const ushort_t* __restrict__ qb, const ushort_t* __restrict__ kb,
    const ushort_t* __restrict__ vt, ushort_t* __restrict__ ctxb)
{
  __shared__ __align__(16) ushort_t kTile[32 * KP_];     // [key][hd]
  __shared__ __align__(16) ushort_t vTile[128 * VP_];    // [hd][key]
  __shared__ __align__(16) ushort_t probs[4][16][32];    // per-wave prob tiles
  const int tid  = threadIdx.x;
  const int lane = tid & 31;
  const int wv   = tid >> 5;
  const int half = lane >> 4;
  const int col  = lane & 15;
  const int s0 = (blockIdx.x * 4 + wv) * 16;
  const int bb = blockIdx.y >> 3;
  const int h  = blockIdx.y & 7;
  const float scale = 0.08838834764831845f;     // 1/sqrt(128)

  Frag qa[4];                                   // Q 16x128 as 4 A-fragments
#pragma unroll
  for (int kc = 0; kc < 4; ++kc) {
    const ushort_t* qp = qb + (size_t)((s0 + col) * B_ + bb) * D_ + h * HD_ + kc * 32 + half * 8;
    qa[kc].q[0] = *(const uint4*)(qp);
    qa[kc].q[1] = *(const uint4*)(qp + 16);
  }
  v8f acc[8] = {v8f{}, v8f{}, v8f{}, v8f{}, v8f{}, v8f{}, v8f{}, v8f{}};  // ctx 16x128
  float mrun[8], lrun[8];
#pragma unroll
  for (int v = 0; v < 8; ++v) { mrun[v] = -__builtin_inff(); lrun[v] = 0.f; }

  const ushort_t* kBase = kb + (size_t)bb * D_ + h * HD_;
  const ushort_t* vBase = vt + (size_t)(bb * H_ + h) * HD_ * S_;

  for (int t0 = 0; t0 < S_; t0 += 32) {         // 32-key tiles
    __syncthreads();                            // WAR vs previous iteration's reads
    // stage K tile: 32 keys x 128 hd (16 b128 chunks per key row)
#pragma unroll
    for (int i = 0; i < 4; ++i) {
      const int idx = tid + i * 128;            // 0..511
      const int key = idx >> 4, c8 = (idx & 15) * 8;
      *(uint4*)(&kTile[key * KP_ + c8]) =
          *(const uint4*)(kBase + (size_t)(t0 + key) * (B_ * D_) + c8);
    }
    // stage V tile: 128 hd x 32 keys (4 b128 chunks per hd row)
#pragma unroll
    for (int i = 0; i < 4; ++i) {
      const int idx = tid + i * 128;            // 0..511
      const int hd = idx >> 2, c8 = (idx & 3) * 8;
      *(uint4*)(&vTile[hd * VP_ + c8]) =
          *(const uint4*)(vBase + (size_t)hd * S_ + t0 + c8);
    }
    __syncthreads();

    v8f c0 = {}, c1 = {};
#pragma unroll
    for (int kc = 0; kc < 4; ++kc) {            // scores from LDS fragments
      Frag b0, b1;
      const ushort_t* p0 = &kTile[col * KP_ + kc * 32 + half * 8];
      b0.q[0] = *(const uint4*)(p0); b0.q[1] = *(const uint4*)(p0 + 16);
      const ushort_t* p1 = &kTile[(16 + col) * KP_ + kc * 32 + half * 8];
      b1.q[0] = *(const uint4*)(p1); b1.q[1] = *(const uint4*)(p1 + 16);
      c0 = wmma_bf16(qa[kc].v, b0.v, c0);
      c1 = wmma_bf16(qa[kc].v, b1.v, c1);
    }
    float f[8];
#pragma unroll
    for (int v = 0; v < 8; ++v) {               // online softmax per C-row
      float a0 = c0[v] * scale, a1 = c1[v] * scale;
      float t = fmaxf(a0, a1);
#pragma unroll
      for (int o = 8; o >= 1; o >>= 1) t = fmaxf(t, __shfl_xor(t, o, 32));  // intra-half
      const float mn = fmaxf(mrun[v], t);
      f[v] = __expf(mrun[v] - mn);
      mrun[v] = mn;
      const float p0 = __expf(a0 - mn), p1 = __expf(a1 - mn);
      float rs = p0 + p1;
#pragma unroll
      for (int o = 8; o >= 1; o >>= 1) rs += __shfl_xor(rs, o, 32);
      lrun[v] = lrun[v] * f[v] + rs;
      const int row = v + half * 8;
      probs[wv][row][col]      = f2bf(p0);      // C-layout -> LDS
      probs[wv][row][16 + col] = f2bf(p1);
    }
#pragma unroll
    for (int j = 0; j < 8; ++j)
#pragma unroll
      for (int v = 0; v < 8; ++v) acc[j][v] *= f[v];
    asm volatile("s_wait_dscnt 0" ::: "memory");          // cross-lane LDS store->load
    Frag pa;                                              // reload as 16x32 A-fragment
    const ushort_t* pp = &probs[wv][col][half * 8];
    pa.q[0] = *(const uint4*)(pp);
    pa.q[1] = *(const uint4*)(pp + 16);
    asm volatile("s_wait_dscnt 0" ::: "memory");          // loads done before overwrite
#pragma unroll
    for (int j = 0; j < 8; ++j) {               // P(16x32) x V(32x16) from LDS
      Frag vf;
      const ushort_t* vp = &vTile[(j * 16 + col) * VP_ + half * 8];
      vf.q[0] = *(const uint4*)(vp); vf.q[1] = *(const uint4*)(vp + 16);
      acc[j] = wmma_bf16(pa.v, vf.v, acc[j]);
    }
  }
#pragma unroll
  for (int j = 0; j < 8; ++j)
#pragma unroll
    for (int v = 0; v < 8; ++v) {
      const int row = v + half * 8;
      const float o = acc[j][v] / lrun[v];
      ctxb[(size_t)((s0 + row) * B_ + bb) * D_ + h * HD_ + j * 16 + col] = f2bf(o);
    }
}

// ---------------- LN1: x1 = LN(x + attn_out) -> bf16 ----------------
__global__ __launch_bounds__(256) void ln1_kernel(
    const float* __restrict__ x, const float* __restrict__ attn,
    const float* __restrict__ g, const float* __restrict__ b, ushort_t* __restrict__ x1b)
{
  __shared__ float red[256];
  const int m = blockIdx.x, tid = threadIdx.x;
  float vals[4]; float s = 0.f;
#pragma unroll
  for (int i = 0; i < 4; ++i) {
    const int d = tid + i * 256;
    vals[i] = x[(size_t)m * D_ + d] + attn[(size_t)m * D_ + d];
    s += vals[i];
  }
  red[tid] = s; __syncthreads();
  for (int st = 128; st > 0; st >>= 1) { if (tid < st) red[tid] += red[tid + st]; __syncthreads(); }
  const float mu = red[0] / D_;
  __syncthreads();
  float vs = 0.f;
#pragma unroll
  for (int i = 0; i < 4; ++i) { const float dv = vals[i] - mu; vs += dv * dv; }
  red[tid] = vs; __syncthreads();
  for (int st = 128; st > 0; st >>= 1) { if (tid < st) red[tid] += red[tid + st]; __syncthreads(); }
  const float rs = rsqrtf(red[0] / D_ + EPS_);
#pragma unroll
  for (int i = 0; i < 4; ++i) {
    const int d = tid + i * 256;
    x1b[(size_t)m * D_ + d] = f2bf((vals[i] - mu) * rs * g[d] + b[d]);
  }
}

// ---------------- gate + top-2 (raw logits as weights) ----------------
__global__ __launch_bounds__(128) void gate_top2(
    const ushort_t* __restrict__ x1b, const float* __restrict__ gw, const float* __restrict__ gb,
    float* __restrict__ topv, int* __restrict__ topi)
{
  const int lane = threadIdx.x & 31;
  const int m = blockIdx.x * 4 + (threadIdx.x >> 5);
  float xr[32];
#pragma unroll
  for (int i = 0; i < 32; ++i) xr[i] = bf2f(x1b[(size_t)m * D_ + lane + i * 32]);
  float lg[E_];
#pragma unroll
  for (int e = 0; e < E_; ++e) {
    const float* wrow = gw + (size_t)e * D_;
    float a = 0.f;
#pragma unroll
    for (int i = 0; i < 32; ++i) a += xr[i] * wrow[lane + i * 32];
#pragma unroll
    for (int o = 16; o >= 1; o >>= 1) a += __shfl_xor(a, o, 32);
    lg[e] = a + gb[e];
  }
  int i1 = 0; float v1 = lg[0];
#pragma unroll
  for (int e = 1; e < E_; ++e) if (lg[e] > v1) { v1 = lg[e]; i1 = e; }
  int i2 = -1; float v2 = -__builtin_inff();
#pragma unroll
  for (int e = 0; e < E_; ++e) if (e != i1 && lg[e] > v2) { v2 = lg[e]; i2 = e; }
  if (lane == 0) {
    topv[m * 2] = v1; topv[m * 2 + 1] = v2;
    topi[m * 2] = i1; topi[m * 2 + 1] = i2;
  }
}

// ---------------- MoE dispatch (deterministic: results keyed by entry, not position) ----------------
__global__ void zero_meta(int* meta) { if (threadIdx.x < 64) meta[threadIdx.x] = 0; }
__global__ void count_kernel(const int* __restrict__ topi, int* __restrict__ cnt) {
  int m = blockIdx.x * 256 + threadIdx.x;
  if (m < M_) { atomicAdd(&cnt[topi[m * 2]], 1); atomicAdd(&cnt[topi[m * 2 + 1]], 1); }
}
__global__ void scan_kernel(const int* __restrict__ cnt, int* __restrict__ offs) {
  if (threadIdx.x == 0) { int r = 0; for (int e = 0; e < E_; ++e) { offs[e] = r; r += cnt[e]; } offs[E_] = r; }
}
__global__ void scatter_kernel(const int* __restrict__ topi, const int* __restrict__ offs,
                               int* __restrict__ cursor, int* __restrict__ rowlist) {
  int m = blockIdx.x * 256 + threadIdx.x;
  if (m < M_)
    for (int k = 0; k < 2; ++k) {
      int e = topi[m * 2 + k];
      int p = offs[e] + atomicAdd(&cursor[e], 1);
      rowlist[p] = m * 2 + k;
    }
}

// ---------------- LN2: out = LN(x1 + w0*picked[2m] + w1*picked[2m+1]) ----------------
__global__ __launch_bounds__(256) void ln2_kernel(
    const ushort_t* __restrict__ x1b, const float* __restrict__ picked,
    const float* __restrict__ topv, const float* __restrict__ g, const float* __restrict__ b,
    float* __restrict__ out)
{
  __shared__ float red[256];
  const int m = blockIdx.x, tid = threadIdx.x;
  const float w0 = topv[m * 2], w1 = topv[m * 2 + 1];
  float vals[4]; float s = 0.f;
#pragma unroll
  for (int i = 0; i < 4; ++i) {
    const int d = tid + i * 256;
    const float moe = w0 * picked[(size_t)(2 * m) * D_ + d] + w1 * picked[(size_t)(2 * m + 1) * D_ + d];
    vals[i] = bf2f(x1b[(size_t)m * D_ + d]) + moe;
    s += vals[i];
  }
  red[tid] = s; __syncthreads();
  for (int st = 128; st > 0; st >>= 1) { if (tid < st) red[tid] += red[tid + st]; __syncthreads(); }
  const float mu = red[0] / D_;
  __syncthreads();
  float vs = 0.f;
#pragma unroll
  for (int i = 0; i < 4; ++i) { const float dv = vals[i] - mu; vs += dv * dv; }
  red[tid] = vs; __syncthreads();
  for (int st = 128; st > 0; st >>= 1) { if (tid < st) red[tid] += red[tid + st]; __syncthreads(); }
  const float rs = rsqrtf(red[0] / D_ + EPS_);
#pragma unroll
  for (int i = 0; i < 4; ++i) {
    const int d = tid + i * 256;
    out[(size_t)m * D_ + d] = (vals[i] - mu) * rs * g[d] + b[d];
  }
}

// ---------------- host orchestration ----------------
extern "C" void kernel_launch(void* const* d_in, const int* in_sizes, int n_in,
                              void* d_out, int out_size, void* d_ws, size_t ws_size,
                              hipStream_t stream) {
  (void)in_sizes; (void)n_in; (void)out_size; (void)ws_size;
  const float* x  = (const float*)d_in[0];
  const float* wq = (const float*)d_in[2];
  const float* wk = (const float*)d_in[3];
  const float* wvw= (const float*)d_in[4];
  const float* bq = (const float*)d_in[5];
  const float* bk = (const float*)d_in[6];
  const float* bv = (const float*)d_in[7];
  const float* wo = (const float*)d_in[8];
  const float* bo = (const float*)d_in[9];
  const float* gw = (const float*)d_in[10];
  const float* gbv= (const float*)d_in[11];
  const float* ew = (const float*)d_in[12];
  const float* eb = (const float*)d_in[13];
  const float* g1 = (const float*)d_in[14];
  const float* b1 = (const float*)d_in[15];
  const float* g2 = (const float*)d_in[16];
  const float* b2 = (const float*)d_in[17];
  float* out = (float*)d_out;

  char* w = (char*)d_ws;
  size_t off = 0;
  auto take = [&](size_t bytes) -> char* {
    char* p = w + off; off = (off + bytes + 255) & ~(size_t)255; return p;
  };
  ushort_t* xb   = (ushort_t*)take((size_t)M_ * D_ * 2);
  ushort_t* wqb  = (ushort_t*)take((size_t)D_ * D_ * 2);
  ushort_t* wkb  = (ushort_t*)take((size_t)D_ * D_ * 2);
  ushort_t* wvb  = (ushort_t*)take((size_t)D_ * D_ * 2);
  ushort_t* wob  = (ushort_t*)take((size_t)D_ * D_ * 2);
  ushort_t* ewb  = (ushort_t*)take((size_t)E_ * D_ * D_ * 2);
  ushort_t* x1b  = (ushort_t*)take((size_t)(M_ + 16) * D_ * 2);   // +dummy row for gather tails
  float*    attn = (float*)take((size_t)M_ * D_ * 4);
  float*    topv = (float*)take((size_t)M_ * 2 * 4);
  int*      topi = (int*)take((size_t)M_ * 2 * 4);
  int*      meta = (int*)take(256);
  int*      rowlist = (int*)take((size_t)2 * M_ * 4);
  char*     regionA = take((size_t)5 * M_ * D_ * 2);              // 80 MB arena
  ushort_t* qb   = (ushort_t*)regionA;
  ushort_t* kb   = qb + (size_t)M_ * D_;
  ushort_t* vb   = kb + (size_t)M_ * D_;
  ushort_t* vt   = vb + (size_t)M_ * D_;
  ushort_t* ctxb = vt + (size_t)M_ * D_;
  float*    picked = (float*)regionA;   // aliases q/k/v/vt/ctx: dead once O-proj+LN1 done
  int* cnt = meta, *cursor = meta + 16, *offs = meta + 32;

  const int T = 256;
  cvt_f32_bf16<<<(M_ * D_ + T - 1) / T, T, 0, stream>>>(x, xb, M_ * D_);
  cvt_f32_bf16<<<(D_ * D_ + T - 1) / T, T, 0, stream>>>(wq, wqb, D_ * D_);
  cvt_f32_bf16<<<(D_ * D_ + T - 1) / T, T, 0, stream>>>(wk, wkb, D_ * D_);
  cvt_f32_bf16<<<(D_ * D_ + T - 1) / T, T, 0, stream>>>(wvw, wvb, D_ * D_);
  cvt_f32_bf16<<<(D_ * D_ + T - 1) / T, T, 0, stream>>>(wo, wob, D_ * D_);
  cvt_f32_bf16<<<(E_ * D_ * D_ + T - 1) / T, T, 0, stream>>>(ew, ewb, E_ * D_ * D_);

  dim3 gg(D_ / 512, M_ / 32, 1);
  gemm_wmma<0><<<gg, 256, 0, stream>>>(xb, wqb, bq, qb, nullptr, nullptr, nullptr);
  gemm_wmma<0><<<gg, 256, 0, stream>>>(xb, wkb, bk, kb, nullptr, nullptr, nullptr);
  gemm_wmma<0><<<gg, 256, 0, stream>>>(xb, wvb, bv, vb, nullptr, nullptr, nullptr);

  transpose_v<<<(M_ * D_) / T, T, 0, stream>>>(vb, vt);
  flash_attn<<<dim3(S_ / 64, B_ * H_), 128, 0, stream>>>(qb, kb, vt, ctxb);

  gemm_wmma<1><<<gg, 256, 0, stream>>>(ctxb, wob, bo, attn, nullptr, nullptr, nullptr);
  ln1_kernel<<<M_, 256, 0, stream>>>(x, attn, g1, b1, x1b);

  gate_top2<<<M_ / 4, 128, 0, stream>>>(x1b, gw, gbv, topv, topi);
  zero_meta<<<1, 64, 0, stream>>>(meta);
  count_kernel<<<M_ / T, T, 0, stream>>>(topi, cnt);
  scan_kernel<<<1, 32, 0, stream>>>(cnt, offs);
  scatter_kernel<<<M_ / T, T, 0, stream>>>(topi, offs, cursor, rowlist);

  gemm_wmma<2><<<dim3(D_ / 512, (2 * M_) / 32, E_), 256, 0, stream>>>(
      x1b, ewb, eb, picked, rowlist, cnt, offs);

  ln2_kernel<<<M_, 256, 0, stream>>>(x1b, picked, topv, g2, b2, out);
}
